// HeteroGCN_9448928051239
// MI455X (gfx1250) — compile-verified, matrix operand order checked by reading, not run on hardware
//
#include <hip/hip_runtime.h>

typedef __attribute__((ext_vector_type(16))) _Float16 v16h;
typedef __attribute__((ext_vector_type(8)))  _Float16 v8h;
typedef __attribute__((ext_vector_type(4)))  _Float16 v4h;
typedef __attribute__((ext_vector_type(8)))  float    v8f;

#define LN_EPS 1e-5f

// ---------------- utility kernels ----------------

__global__ void zero_f32(float* __restrict__ p, int n) {
    int i = blockIdx.x * 256 + threadIdx.x;
    if (i < n) p[i] = 0.0f;
}

__global__ void f32_to_f16(const float* __restrict__ s, _Float16* __restrict__ d, int n) {
    int i = blockIdx.x * 256 + threadIdx.x;
    if (i < n) d[i] = (_Float16)s[i];
}

// gene receives dg+gg+cg: sum the three root weights (to f16) and the three biases (f32)
__global__ void combine_gene(const float* __restrict__ wr_dg, const float* __restrict__ wr_gg,
                             const float* __restrict__ wr_cg,
                             const float* __restrict__ bl_dg, const float* __restrict__ bl_gg,
                             const float* __restrict__ bl_cg,
                             _Float16* __restrict__ wrs, float* __restrict__ bsum) {
    int i = blockIdx.x * 256 + threadIdx.x;
    if (i < 128 * 128) wrs[i] = (_Float16)(wr_dg[i] + wr_gg[i] + wr_cg[i]);
    if (i < 128)       bsum[i] = bl_dg[i] + bl_gg[i] + bl_cg[i];
}

__global__ void count_edges(const int* __restrict__ dst, int E, float* __restrict__ cnt) {
    int e = blockIdx.x * 256 + threadIdx.x;
    if (e < E) atomicAdd(&cnt[dst[e]], 1.0f);
}

// convert counts to mean-scale reciprocals once: cnt[i] <- 1/max(cnt[i],1)
__global__ void recip_counts(float* __restrict__ c, int n) {
    int i = blockIdx.x * 256 + threadIdx.x;
    if (i < n) c[i] = 1.0f / fmaxf(c[i], 1.0f);
}

// one wave per edge; lane covers 4 consecutive features (32 lanes * 4 = 128)
__global__ void scatter_edges(const float* __restrict__ h, const int* __restrict__ src,
                              const int* __restrict__ dst, int E, float* __restrict__ agg) {
    int gid  = blockIdx.x * 256 + threadIdx.x;
    int e    = gid >> 5;
    int lane = threadIdx.x & 31;
    if (e >= E) return;
    int s = src[e], d = dst[e];
    const float4 v = *reinterpret_cast<const float4*>(h + (size_t)s * 128u + lane * 4);
    float* b = agg + (size_t)d * 128u + lane * 4;
    atomicAdd(b + 0, v.x);
    atomicAdd(b + 1, v.y);
    atomicAdd(b + 2, v.z);
    atomicAdd(b + 3, v.w);
}

// ---------------- WMMA GEMM ----------------
// out[M x 128] (+)= f16(A .* rscale) @ f16(W)[128 x K].T  (+ bias / + relu / + layernorm)
// Block: 256 threads = 8 waves; block covers 16 rows x 128 cols; wave w owns cols [16w,16w+16).
// A tile (16x32 f32) is staged to a double-buffered f16 LDS tile by the first 128 threads
// (float4 global load -> packed 4xf16 LDS store); one barrier per k-step, stage of tile i+1
// overlaps WMMA of tile i. W is pre-converted f16; row n of W == column n of B.
template <int K, int ACCUM, int RELU, int LNORM, int SCALE>
__global__ __launch_bounds__(256) void wmma_gemm(
    const float* __restrict__ A, int M,
    const _Float16* __restrict__ Wh,            // [128][K], f16
    const float* __restrict__ bias,             // [128] or nullptr (ignored if ACCUM)
    const float* __restrict__ rscale,           // [M] reciprocal scales (SCALE=1)
    const float* __restrict__ ln_g, const float* __restrict__ ln_b,
    float* __restrict__ out) {
    __shared__ _Float16 lds_a[2][16][32];
    __shared__ float    lds_o[LNORM ? 16 * 136 : 1];

    const int tid  = threadIdx.x;
    const int lane = tid & 31;
    const int wave = tid >> 5;
    const int l16  = lane & 15;
    const int hi   = lane >> 4;          // 0 or 1 (half-wave)
    const int row0 = blockIdx.x * 16;
    const int col  = wave * 16 + l16;    // output column (0..127)

    // staging: threads 0..127, each owns (row tid>>3, 4 cols at (tid&7)*4) of the 16x32 tile
    const bool stager = tid < 128;
    const int  str = tid >> 3;           // 0..15
    const int  scg = (tid & 7) * 4;      // 0,4,...,28
    int gr = row0 + str; if (gr >= M) gr = M - 1;
    const float* arow = A + (size_t)gr * (size_t)K;
    const float sc = SCALE ? rscale[gr] : 1.0f;

    auto stage = [&](int it, int buf) {
        if (stager) {
            float4 v = *reinterpret_cast<const float4*>(arow + it * 32 + scg);
            if (SCALE) { v.x *= sc; v.y *= sc; v.z *= sc; v.w *= sc; }
            v4h h4;
            h4[0] = (_Float16)v.x; h4[1] = (_Float16)v.y;
            h4[2] = (_Float16)v.z; h4[3] = (_Float16)v.w;
            *reinterpret_cast<v4h*>(&lds_a[buf][str][scg]) = h4;
        }
    };

    v8f acc;
    if (ACCUM) {
#pragma unroll
        for (int j = 0; j < 8; ++j) {
            int r = row0 + j + hi * 8;
            if (r >= M) r = M - 1;
            acc[j] = out[(size_t)r * 128u + col];
        }
    } else {
        float bv = bias ? bias[col] : 0.0f;
#pragma unroll
        for (int j = 0; j < 8; ++j) acc[j] = bv;
    }

    constexpr int NIT = K / 32;
    stage(0, 0);
#pragma unroll
    for (int it = 0; it < NIT; ++it) {
        __syncthreads();                        // buf it&1 ready; prior reads of it&1^1 done
        if (it + 1 < NIT) stage(it + 1, (it + 1) & 1);

        // A fragment: lanes 0-15 row M=l16 K={0..7,16..23}; lanes 16-31 K={8..15,24..31}
        const int koff = hi * 8;
        v8h a0 = *reinterpret_cast<const v8h*>(&lds_a[it & 1][l16][koff]);
        v8h a1 = *reinterpret_cast<const v8h*>(&lds_a[it & 1][l16][16 + koff]);
        // B fragment: column n of B = row n of W (contiguous), same K pattern
        const _Float16* wrow = Wh + (size_t)col * (size_t)K + it * 32;
        v8h b0 = *reinterpret_cast<const v8h*>(wrow + koff);
        v8h b1 = *reinterpret_cast<const v8h*>(wrow + 16 + koff);
        v16h af, bf;
#pragma unroll
        for (int i = 0; i < 8; ++i) {
            af[i] = a0[i]; af[8 + i] = a1[i];
            bf[i] = b0[i]; bf[8 + i] = b1[i];
        }
        acc = __builtin_amdgcn_wmma_f32_16x16x32_f16(false, af, false, bf,
                                                     (short)0, acc, false, false);
    }

    if (LNORM) {
        // cross-wave LayerNorm over the full 128-wide row
        __syncthreads();
#pragma unroll
        for (int j = 0; j < 8; ++j) lds_o[(j + hi * 8) * 136 + col] = acc[j];
        __syncthreads();
        const int row = tid >> 4, sub = tid & 15;  // 16 threads per row (wave32-aligned halves)
        float s = 0.0f, s2 = 0.0f;
#pragma unroll
        for (int c = 0; c < 8; ++c) {
            float v = lds_o[row * 136 + sub * 8 + c];
            s += v; s2 += v * v;
        }
#pragma unroll
        for (int off = 1; off < 16; off <<= 1) {
            s  += __shfl_xor(s,  off, 32);
            s2 += __shfl_xor(s2, off, 32);
        }
        float mu   = s * (1.0f / 128.0f);
        float var  = s2 * (1.0f / 128.0f) - mu * mu;
        float rstd = rsqrtf(var + LN_EPS);
        int grr = row0 + row;
        if (grr < M) {
#pragma unroll
            for (int c = 0; c < 8; ++c) {
                int cc = sub * 8 + c;
                out[(size_t)grr * 128u + cc] =
                    (lds_o[row * 136 + cc] - mu) * rstd * ln_g[cc] + ln_b[cc];
            }
        }
    } else {
#pragma unroll
        for (int j = 0; j < 8; ++j) {
            float v = acc[j];
            if (RELU) v = fmaxf(v, 0.0f);
            int r = row0 + j + hi * 8;
            if (r < M) out[(size_t)r * 128u + col] = v;
        }
    }
}

// ---------------- host launcher ----------------

static inline int cdiv(int a, int b) { return (a + b - 1) / b; }

extern "C" void kernel_launch(void* const* d_in, const int* in_sizes, int n_in,
                              void* d_out, int out_size, void* d_ws, size_t ws_size,
                              hipStream_t stream) {
    (void)in_sizes; (void)n_in; (void)out_size; (void)ws_size;

    // -------- input map (top-level dict order; params = jax tree leaves, dict keys sorted) ----
    const float* xs[3] = {(const float*)d_in[0], (const float*)d_in[1], (const float*)d_in[2]};
    const int* esrc[5]; const int* edst[5];
    for (int i = 0; i < 5; ++i) { esrc[i] = (const int*)d_in[3 + 2 * i]; edst[i] = (const int*)d_in[4 + 2 * i]; }
    // params: 13 + l*15 + slot*3 + {Wl,bl,Wr}; slots sorted: cg=0,dg=1,gc=2,gd=3,gg=4
    const float* ln_g = (const float*)d_in[43];
    const float* ln_b = (const float*)d_in[44];
    const float* pW[3] = {(const float*)d_in[47], (const float*)d_in[49], (const float*)d_in[45]}; // drug,gene,cell
    const float* pB[3] = {(const float*)d_in[48], (const float*)d_in[50], (const float*)d_in[46]};

    // edge-type tables (order: dg, gd, gg, gc, cg)
    static const int EK_E[5]    = {200000, 200000, 600000, 300000, 300000};
    static const int EK_SROW[5] = {0,      8000,   8000,   8000,   48000};  // src row offset in h
    static const int EK_DROW[5] = {8000,   0,      8000,   48000,  8000};   // dst row offset in h
    static const int EK_NDST[5] = {40000,  8000,   40000,  15000,  40000};
    static const int EK_CNT[5]  = {0,      40000,  48000,  88000,  103000};
    static const int EK_SLOT[5] = {1,      3,      4,      2,      0};      // -> sorted param slot
    static const int EK_RELU[5] = {0,      1,      0,      1,      1};      // last GEMM per dst type

    // -------- workspace layout (floats) --------
    float* ws   = (float*)d_ws;
    float* HC   = ws;                  // 63000*128
    float* HN   = ws + 8064000;        // 63000*128
    float* AGG  = ws + 16128000;       // 40000*128 (reused per edge type)
    float* CNT  = ws + 21248000;       // 143000 (holds reciprocals after recip_counts)
    float* BSUM = ws + 21391000;       // 2*128
    _Float16* hb = (_Float16*)(ws + 21391360);
    _Float16* WPH[3] = {hb, hb + 131072, hb + 163840};      // proj weights f16: drug,gene,cell
    _Float16* Lh[2]  = {hb + 229376, hb + 229376 + 131072}; // per-layer f16 block
    // within Lh[l]: slot*16384 -> Wl(cg,dg,gc,gd,gg); +81920 Wr_gd; +98304 Wr_gc; +114688 Wr_sum_gene

    // -------- 1. convert weights to f16, build gene root-sum --------
    f32_to_f16<<<cdiv(131072, 256), 256, 0, stream>>>(pW[0], WPH[0], 131072);
    f32_to_f16<<<cdiv(32768, 256), 256, 0, stream>>>(pW[1], WPH[1], 32768);
    f32_to_f16<<<cdiv(65536, 256), 256, 0, stream>>>(pW[2], WPH[2], 65536);
    for (int l = 0; l < 2; ++l) {
        int base = 13 + l * 15;
        for (int slot = 0; slot < 5; ++slot)
            f32_to_f16<<<64, 256, 0, stream>>>((const float*)d_in[base + slot * 3], Lh[l] + slot * 16384, 16384);
        f32_to_f16<<<64, 256, 0, stream>>>((const float*)d_in[base + 3 * 3 + 2], Lh[l] + 81920, 16384); // Wr_gd
        f32_to_f16<<<64, 256, 0, stream>>>((const float*)d_in[base + 2 * 3 + 2], Lh[l] + 98304, 16384); // Wr_gc
        combine_gene<<<64, 256, 0, stream>>>(
            (const float*)d_in[base + 1 * 3 + 2], (const float*)d_in[base + 4 * 3 + 2],
            (const float*)d_in[base + 0 * 3 + 2],
            (const float*)d_in[base + 1 * 3 + 1], (const float*)d_in[base + 4 * 3 + 1],
            (const float*)d_in[base + 0 * 3 + 1],
            Lh[l] + 114688, BSUM + l * 128);
    }

    // -------- 2. per-edge-type in-degree counts -> reciprocals (shared by both layers) --------
    zero_f32<<<cdiv(143000, 256), 256, 0, stream>>>(CNT, 143000);
    for (int i = 0; i < 5; ++i)
        count_edges<<<cdiv(EK_E[i], 256), 256, 0, stream>>>(edst[i], EK_E[i], CNT + EK_CNT[i]);
    recip_counts<<<cdiv(143000, 256), 256, 0, stream>>>(CNT, 143000);

    // -------- 3. input projection + LayerNorm -> HC --------
    wmma_gemm<1024, 0, 0, 1, 0><<<cdiv(8000, 16), 256, 0, stream>>>(
        xs[0], 8000, WPH[0], pB[0], nullptr, ln_g, ln_b, HC);
    wmma_gemm<256, 0, 0, 1, 0><<<cdiv(40000, 16), 256, 0, stream>>>(
        xs[1], 40000, WPH[1], pB[1], nullptr, ln_g, ln_b, HC + (size_t)8000 * 128);
    wmma_gemm<512, 0, 0, 1, 0><<<cdiv(15000, 16), 256, 0, stream>>>(
        xs[2], 15000, WPH[2], pB[2], nullptr, ln_g, ln_b, HC + (size_t)48000 * 128);

    // -------- 4. two SAGE layers --------
    for (int l = 0; l < 2; ++l) {
        int base = 13 + l * 15;
        const float* hin = (l == 0) ? HC : HN;
        float* hout = (l == 0) ? HN : (float*)d_out;

        // root GEMMs (initialize hout with summed bias as C)
        wmma_gemm<128, 0, 0, 0, 0><<<cdiv(40000, 16), 256, 0, stream>>>(       // gene
            hin + (size_t)8000 * 128, 40000, Lh[l] + 114688, BSUM + l * 128,
            nullptr, nullptr, nullptr, hout + (size_t)8000 * 128);
        wmma_gemm<128, 0, 0, 0, 0><<<cdiv(8000, 16), 256, 0, stream>>>(        // drug (gd)
            hin, 8000, Lh[l] + 81920, (const float*)d_in[base + 3 * 3 + 1],
            nullptr, nullptr, nullptr, hout);
        wmma_gemm<128, 0, 0, 0, 0><<<cdiv(15000, 16), 256, 0, stream>>>(       // cell (gc)
            hin + (size_t)48000 * 128, 15000, Lh[l] + 98304, (const float*)d_in[base + 2 * 3 + 1],
            nullptr, nullptr, nullptr, hout + (size_t)48000 * 128);

        // neighbor aggregation + Wl GEMM per edge type (serialized; agg buffer reused)
        for (int i = 0; i < 5; ++i) {
            int nAgg = EK_NDST[i] * 128;
            zero_f32<<<cdiv(nAgg, 256), 256, 0, stream>>>(AGG, nAgg);
            scatter_edges<<<cdiv(EK_E[i] * 32, 256), 256, 0, stream>>>(
                hin + (size_t)EK_SROW[i] * 128, esrc[i], edst[i], EK_E[i], AGG);
            const _Float16* wl = Lh[l] + EK_SLOT[i] * 16384;
            float* o = hout + (size_t)EK_DROW[i] * 128;
            if (l == 0 && EK_RELU[i])
                wmma_gemm<128, 1, 1, 0, 1><<<cdiv(EK_NDST[i], 16), 256, 0, stream>>>(
                    AGG, EK_NDST[i], wl, nullptr, CNT + EK_CNT[i], nullptr, nullptr, o);
            else
                wmma_gemm<128, 1, 0, 0, 1><<<cdiv(EK_NDST[i], 16), 256, 0, stream>>>(
                    AGG, EK_NDST[i], wl, nullptr, CNT + EK_CNT[i], nullptr, nullptr, o);
        }
    }
}
